// MultiVectorQuantizerEMA_76682346102952
// MI455X (gfx1250) — matrix operand order repaired and checked by reading, not compiled, not run on hardware
//
#include <hip/hip_runtime.h>
#include <hip/hip_bf16.h>
#include <math.h>

#define K_CODES 1024
#define GD      128
#define NGRP    8
#define NROWS   32768
#define DECAYF  0.99f

typedef __attribute__((ext_vector_type(16))) __bf16 v16bf;
typedef __attribute__((ext_vector_type(8)))  __bf16 v8bf;
typedef __attribute__((ext_vector_type(8)))  float  v8f;

// ---------------------------------------------------------------------------
// prep: copy input codebook to fp32+bf16 working copies, compute 0.5*||e||^2,
// zero EMA state and scalar accumulators. Re-run every call (deterministic).
// ---------------------------------------------------------------------------
__global__ __launch_bounds__(GD) void vq_prep(const float* __restrict__ emb_in,
                                              float* __restrict__ emb_f32,
                                              __bf16* __restrict__ emb_bf16,
                                              float* __restrict__ en2,
                                              float* __restrict__ hidden_cs,
                                              float* __restrict__ hidden_dw,
                                              float* __restrict__ scalars) {
  const int k = blockIdx.x;
  const int j = threadIdx.x;
  const int i = k * GD + j;
  float v = emb_in[i];
  emb_f32[i]  = v;
  emb_bf16[i] = (__bf16)v;
  hidden_dw[i] = 0.0f;
  float acc = v * v;
  #pragma unroll
  for (int d = 16; d > 0; d >>= 1) acc += __shfl_xor(acc, d);
  __shared__ float sm[4];
  const int lane = j & 31, wave = j >> 5;
  if (lane == 0) sm[wave] = acc;
  __syncthreads();
  if (j == 0) {
    en2[k] = 0.5f * (sm[0] + sm[1] + sm[2] + sm[3]);
    hidden_cs[k] = 0.0f;
    if (k == 0) { scalars[0] = 0.0f; scalars[1] = 0.0f; }
  }
}

__global__ void vq_zero(float* __restrict__ cs, float* __restrict__ dw) {
  const int i = blockIdx.x * blockDim.x + threadIdx.x;
  if (i < K_CODES * GD) dw[i] = 0.0f;
  if (i < K_CODES)      cs[i] = 0.0f;
}

// ---------------------------------------------------------------------------
// assign: per wave, one 16-row tile of blk. D = emb(bf16,16codes x 32gd) @
// blk(bf16, 32gd x 16rows) accumulated over GD=128 with C preloaded to
// -0.5||e||^2, so argmin(dist) == argmax(D). Each lane tracks best over its
// 8 code-VGPRs x 64 code tiles; halves merged with shfl_xor(16).
// Phase 2 gathers q = emb_f32[idx], writes quantized, accumulates sq error.
// ---------------------------------------------------------------------------
__global__ __launch_bounds__(256, 1) void vq_assign(
    const float* __restrict__ x, const __bf16* __restrict__ embh,
    const float* __restrict__ emb, const float* __restrict__ en2,
    int* __restrict__ idxbuf, float* __restrict__ outq,
    float* __restrict__ scalars, int g) {
  const int lane    = threadIdx.x & 31;
  const int wave    = threadIdx.x >> 5;
  const int rowbase = (blockIdx.x * 8 + wave) * 16;
  const int col     = lane & 15;
  const int hiHalf  = lane >> 4;          // 0 or 1
  const int row     = rowbase + col;
  const float* xrow = x + (size_t)row * 1024 + g * GD;

  // B fragments: blk as 32(gd) x 16(rows), 4 chunks covering GD=128.
  // Lane holds col=row; halves h=0..15 are K = kc*32 + hiHalf*16 + h.
  v16bf B[4];
  #pragma unroll
  for (int kc = 0; kc < 4; ++kc) {
    const float* xp = xrow + kc * 32 + hiHalf * 16;
    #pragma unroll
    for (int i = 0; i < 16; ++i) B[kc][i] = (__bf16)xp[i];
  }

  float bestVal = -3.4e38f;
  int   bestIdx = 0;
  const int mBase = hiHalf * 8;

  for (int ct = 0; ct < 64; ++ct) {
    // C init: -0.5*||e||^2 for this lane's 8 codes (M = mBase..mBase+7)
    const float* e2 = en2 + ct * 16 + mBase;
    v8f c;
    #pragma unroll
    for (int v = 0; v < 8; ++v) c[v] = -e2[v];
    #pragma unroll
    for (int kc = 0; kc < 4; ++kc) {
      // A: emb tile 16(codes) x 32(gd); lane row = col;
      // lanes 0-15: K={0..15}, lanes 16-31: K={16..31} (+kc*32)
      const __bf16* ap = embh + (size_t)(ct * 16 + col) * GD + kc * 32 + hiHalf * 16;
      v8bf alo = *(const v8bf*)(ap);
      v8bf ahi = *(const v8bf*)(ap + 8);
      v16bf A = __builtin_shufflevector(alo, ahi, 0, 1, 2, 3, 4, 5, 6, 7,
                                        8, 9, 10, 11, 12, 13, 14, 15);
      c = __builtin_amdgcn_wmma_f32_16x16x32_bf16(false, A, false, B[kc],
                                                  (short)0, c, false, false);
    }
    #pragma unroll
    for (int v = 0; v < 8; ++v) {
      float s = c[v];
      if (s > bestVal) { bestVal = s; bestIdx = ct * 16 + mBase + v; }
    }
  }
  // merge the two code-halves of each row (lane n <-> lane n+16)
  float oVal = __shfl_xor(bestVal, 16);
  int   oIdx = __shfl_xor(bestIdx, 16);
  if (oVal > bestVal || (oVal == bestVal && oIdx < bestIdx)) {
    bestVal = oVal; bestIdx = oIdx;
  }
  if (lane < 16) idxbuf[row] = bestIdx;

  // Phase 2: quantized output (q = emb_f32[idx]) + latent-loss partial sums
  float acc = 0.0f;
  for (int r = 0; r < 16; ++r) {
    const int k  = __shfl(bestIdx, r);
    const int rr = rowbase + r;
    const float* er = emb + (size_t)k * GD;
    const float* xr = x + (size_t)rr * 1024 + g * GD;
    float* orow     = outq + (size_t)rr * 1024 + g * GD;
    #pragma unroll
    for (int it = 0; it < 4; ++it) {
      const int j = lane + it * 32;
      float qv = er[j];
      float xv = xr[j];
      orow[j] = qv;
      float d = xv - qv;
      acc += d * d;
    }
  }
  #pragma unroll
  for (int d = 16; d > 0; d >>= 1) acc += __shfl_xor(acc, d);
  if (lane == 0) atomicAdd(scalars + 0, acc);
}

// ---------------------------------------------------------------------------
// scatter: cs histogram + dw segment sums via fp32 global atomics (L2-resident)
// ---------------------------------------------------------------------------
__global__ void vq_scatter(const float* __restrict__ x, const int* __restrict__ idx,
                           float* __restrict__ cs, float* __restrict__ dw, int g) {
  const int t = blockIdx.x * blockDim.x + threadIdx.x;
  const int n = t >> 7, j = t & 127;
  const int k = idx[n];
  atomicAdd(&dw[k * GD + j], x[(size_t)n * 1024 + g * GD + j]);
  if (j == 0) atomicAdd(&cs[k], 1.0f);
}

// ---------------------------------------------------------------------------
// perp: perplexity of this group's code usage, accumulated into scalars[1]
// ---------------------------------------------------------------------------
__global__ __launch_bounds__(K_CODES) void vq_perp(const float* __restrict__ cs,
                                                   float* __restrict__ scalars) {
  const int k = threadIdx.x;
  const float p = cs[k] * (1.0f / (float)NROWS);
  float acc = p * logf(p + 1e-6f);
  #pragma unroll
  for (int d = 16; d > 0; d >>= 1) acc += __shfl_xor(acc, d);
  __shared__ float sm[32];
  const int lane = k & 31, wave = k >> 5;
  if (lane == 0) sm[wave] = acc;
  __syncthreads();
  if (wave == 0) {
    float v = sm[lane];
    #pragma unroll
    for (int d = 16; d > 0; d >>= 1) v += __shfl_xor(v, d);
    if (lane == 0) atomicAdd(scalars + 1, expf(-v));
  }
}

// ---------------------------------------------------------------------------
// update: EMA state + new codebook (fp32 and bf16) + 0.5||e||^2.
// n_tot == N exactly (sum(hidden_cs) = N(1-DECAY^c) cancels bias correction).
// ---------------------------------------------------------------------------
__global__ __launch_bounds__(GD) void vq_update(
    const float* __restrict__ cs, const float* __restrict__ dw,
    float* __restrict__ hidden_cs, float* __restrict__ hidden_dw,
    float* __restrict__ emb_f32, __bf16* __restrict__ emb_bf16,
    float* __restrict__ en2, float corr) {
  const int k = blockIdx.x;
  const int j = threadIdx.x;
  const int i = k * GD + j;
  const float hc = hidden_cs[k] * DECAYF + (1.0f - DECAYF) * cs[k];
  const float avg_cs = hc * corr;
  const float upd = (avg_cs + 1e-5f) * ((float)NROWS / ((float)NROWS + (float)K_CODES * 1e-5f));
  const float hd = hidden_dw[i] * DECAYF + (1.0f - DECAYF) * dw[i];
  hidden_dw[i] = hd;
  const float e = (hd * corr) / upd;
  emb_f32[i]  = e;
  emb_bf16[i] = (__bf16)e;
  float acc = e * e;
  #pragma unroll
  for (int d = 16; d > 0; d >>= 1) acc += __shfl_xor(acc, d);
  __shared__ float sm[4];
  const int lane = j & 31, wave = j >> 5;
  if (lane == 0) sm[wave] = acc;
  __syncthreads();
  if (j == 0) {
    en2[k] = 0.5f * (sm[0] + sm[1] + sm[2] + sm[3]);
    hidden_cs[k] = hc;
  }
}

__global__ void vq_finalize(const float* __restrict__ scalars, float* __restrict__ out) {
  const size_t nq = (size_t)NROWS * 1024;
  out[nq]     = 0.25f * scalars[0] / ((float)NROWS * (float)GD * (float)NGRP);
  out[nq + 1] = scalars[1] / (float)NGRP;
}

extern "C" void kernel_launch(void* const* d_in, const int* in_sizes, int n_in,
                              void* d_out, int out_size, void* d_ws, size_t ws_size,
                              hipStream_t stream) {
  const float* x      = (const float*)d_in[0];
  const float* emb_in = (const float*)d_in[1];
  float* out = (float*)d_out;

  char* w = (char*)d_ws;
  auto al = [](size_t o) { return (o + 255) & ~(size_t)255; };
  size_t o = 0;
  float*  emb_f32   = (float*)(w + o);  o = al(o + (size_t)K_CODES * GD * 4);
  __bf16* emb_bf16  = (__bf16*)(w + o); o = al(o + (size_t)K_CODES * GD * 2);
  float*  en2       = (float*)(w + o);  o = al(o + (size_t)K_CODES * 4);
  float*  hidden_cs = (float*)(w + o);  o = al(o + (size_t)K_CODES * 4);
  float*  hidden_dw = (float*)(w + o);  o = al(o + (size_t)K_CODES * GD * 4);
  float*  cs        = (float*)(w + o);  o = al(o + (size_t)K_CODES * 4);
  float*  dw        = (float*)(w + o);  o = al(o + (size_t)K_CODES * GD * 4);
  int*    idxb      = (int*)(w + o);    o = al(o + (size_t)NROWS * 4);
  float*  scalars   = (float*)(w + o);  o = al(o + 2 * 4);

  vq_prep<<<K_CODES, GD, 0, stream>>>(emb_in, emb_f32, emb_bf16, en2,
                                      hidden_cs, hidden_dw, scalars);
  for (int g = 0; g < NGRP; ++g) {
    vq_zero<<<(K_CODES * GD) / 256, 256, 0, stream>>>(cs, dw);
    vq_assign<<<NROWS / 128, 256, 0, stream>>>(x, emb_bf16, emb_f32, en2,
                                               idxb, out, scalars, g);
    vq_scatter<<<(NROWS * GD) / 256, 256, 0, stream>>>(x, idxb, cs, dw, g);
    vq_perp<<<1, K_CODES, 0, stream>>>(cs, scalars);
    const float corr = 1.0f / (1.0f - powf(DECAYF, (float)(g + 1)));
    vq_update<<<K_CODES, GD, 0, stream>>>(cs, dw, hidden_cs, hidden_dw,
                                          emb_f32, emb_bf16, en2, corr);
  }
  vq_finalize<<<1, 1, 0, stream>>>(scalars, out);
}
